// GraphMAE_5377299054918
// MI455X (gfx1250) — compile-verified, matrix operand order checked by reading, not run on hardware
//
#include <hip/hip_runtime.h>

typedef __attribute__((ext_vector_type(16))) __bf16 v16bf;
typedef __attribute__((ext_vector_type(8)))  __bf16 v8bf;
typedef __attribute__((ext_vector_type(8)))  float  v8f;
typedef __attribute__((ext_vector_type(4)))  float  v4f;

#define IN_CH 128
#define HID   64

// ---------------------------------------------------------------------------
// WMMA wrapper: D(16x16 f32) = A(16x32 bf16) * B(32x16 bf16) + C
// ---------------------------------------------------------------------------
__device__ __forceinline__ v8f wmma_bf16(v16bf a, v16bf b, v8f c) {
    return __builtin_amdgcn_wmma_f32_16x16x32_bf16(
        /*neg_a=*/false, a, /*neg_b=*/false, b,
        /*c_mod=*/(short)0, c, /*reuse_a=*/false, /*reuse_b=*/false);
}

__device__ __forceinline__ v8f v8f_zero() {
    v8f z;
#pragma unroll
    for (int i = 0; i < 8; ++i) z[i] = 0.0f;
    return z;
}

// ---------------------------------------------------------------------------
// Zero the aggregation buffer (float4 stores)
// ---------------------------------------------------------------------------
__global__ void k_zero(float* __restrict__ p, int n4) {
    int i = blockIdx.x * blockDim.x + threadIdx.x;
    if (i < n4) {
        v4f z; z[0] = 0.f; z[1] = 0.f; z[2] = 0.f; z[3] = 0.f;
        ((v4f*)p)[i] = z;
    }
}

// ---------------------------------------------------------------------------
// Pre-swizzle weights into bf16 WMMA B-operand layout.
// B-operand (32x16 bf16, wave32): lane L holds column N = L&15; half = L>>4;
// element e (0..15): K = kt*32 + (e<8 ? 8*half+e : 16+8*half+(e-8)).
// sw element layout: [0,8192) W_rel, [8192,16384) W_root, [16384,24576) W_dec
// block of (kt,nt): 32 lanes x 16 bf16, contiguous.
// ---------------------------------------------------------------------------
__global__ void k_prep(const float* __restrict__ Wrel,
                       const float* __restrict__ Wroot,
                       const float* __restrict__ Wdec,
                       __bf16* __restrict__ sw) {
    int lane = threadIdx.x;
    int b = blockIdx.x;
    const float* W; __bf16* o; int NT, NC;
    if (b < 16)      { W = Wrel;  o = sw;         NT = 4; NC = 64;  }
    else if (b < 32) { W = Wroot; o = sw + 8192;  NT = 4; NC = 64;  b -= 16; }
    else             { W = Wdec;  o = sw + 16384; NT = 8; NC = 128; b -= 32; }
    int kt = b / NT, nt = b % NT;
    int row = lane & 15, half = lane >> 4;
#pragma unroll
    for (int e = 0; e < 16; ++e) {
        int k = kt * 32 + ((e < 8) ? (8 * half + e) : (16 + 8 * half + (e - 8)));
        int n = nt * 16 + row;
        o[((b * 32) + lane) * 16 + e] = (__bf16)W[k * NC + n];
    }
}

// ---------------------------------------------------------------------------
// Edge scatter: agg[dst] += x[src]; one thread = one edge x 4 channels.
// Gathers hit L2 (x is 25.6MB << 192MB L2); atomics resolve in L2.
// ---------------------------------------------------------------------------
__global__ void k_scatter(const float* __restrict__ x,
                          const int* __restrict__ ei,
                          float* __restrict__ agg, int E) {
    unsigned t = blockIdx.x * blockDim.x + threadIdx.x;
    unsigned e = t >> 5;
    if (e >= (unsigned)E) return;
    int c = (int)(t & 31u) << 2;
    int src = ei[e];
    int dst = ei[(unsigned)E + e];
    v4f v = *(const v4f*)(x + (size_t)src * IN_CH + c);
    float* a = agg + (size_t)dst * IN_CH + c;
#pragma unroll
    for (int i = 0; i < 4; ++i)
        __hip_atomic_fetch_add(a + i, v[i], __ATOMIC_RELAXED, __HIP_MEMORY_SCOPE_AGENT);
}

// ---------------------------------------------------------------------------
// A-operand loader from row-major f32 (row base + kt*32 + half*8 already
// applied by caller): two contiguous 8-float runs -> v16bf.
// ---------------------------------------------------------------------------
__device__ __forceinline__ v16bf load_a32(const float* p) {
    v4f f0 = *(const v4f*)(p);
    v4f f1 = *(const v4f*)(p + 4);
    v4f f2 = *(const v4f*)(p + 16);
    v4f f3 = *(const v4f*)(p + 20);
    v16bf a;
#pragma unroll
    for (int i = 0; i < 4; ++i) {
        a[i]      = (__bf16)f0[i];
        a[4 + i]  = (__bf16)f1[i];
        a[8 + i]  = (__bf16)f2[i];
        a[12 + i] = (__bf16)f3[i];
    }
    return a;
}

// ---------------------------------------------------------------------------
// Fused GEMM: per 16-node tile (one wave):
//   h  = relu(agg@W_rel + x@W_root + b_rel)   [16x64]   (32 WMMAs)
//   out = h@W_dec + b_dec                      [16x128]  (16 WMMAs)
// h transits LDS (bf16) to re-layout D(lane=N) -> A(lane=M).
// ---------------------------------------------------------------------------
__global__ void __launch_bounds__(32)
k_gemm(const float* __restrict__ agg, const float* __restrict__ x,
       const __bf16* __restrict__ sw,
       const float* __restrict__ brel, const float* __restrict__ bdec,
       float* __restrict__ out) {
    __shared__ __bf16 hl[16 * HID];

    const int lane = threadIdx.x;
    const int row  = lane & 15;
    const int half = lane >> 4;
    const int tile = blockIdx.x;

    const __bf16* wrel  = sw;
    const __bf16* wroot = sw + 8192;
    const __bf16* wdec  = sw + 16384;

    const float* aggr = agg + (size_t)(tile * 16 + row) * IN_CH;
    const float* xr   = x   + (size_t)(tile * 16 + row) * IN_CH;

    // ---- layer 1: h = agg@W_rel + x@W_root ----
    v8f c1[4];
#pragma unroll
    for (int nt = 0; nt < 4; ++nt) c1[nt] = v8f_zero();

#pragma unroll
    for (int kt = 0; kt < 4; ++kt) {
        v16bf aA = load_a32(aggr + kt * 32 + half * 8);
        v16bf aX = load_a32(xr   + kt * 32 + half * 8);
#pragma unroll
        for (int nt = 0; nt < 4; ++nt) {
            v16bf bR = *(const v16bf*)(wrel  + (size_t)((kt * 4 + nt) * 32 + lane) * 16);
            c1[nt] = wmma_bf16(aA, bR, c1[nt]);
            v16bf bT = *(const v16bf*)(wroot + (size_t)((kt * 4 + nt) * 32 + lane) * 16);
            c1[nt] = wmma_bf16(aX, bT, c1[nt]);
        }
    }

    // bias + relu, stage h in LDS (row-major 16x64 bf16).
    // D layout: VGPR r, lane L -> M = r + 8*half, N = nt*16 + (L&15)
#pragma unroll
    for (int nt = 0; nt < 4; ++nt) {
        float bias = brel[nt * 16 + row];
#pragma unroll
        for (int r = 0; r < 8; ++r) {
            float hv = c1[nt][r] + bias;
            hv = fmaxf(hv, 0.0f);
            hl[(r + 8 * half) * HID + nt * 16 + row] = (__bf16)hv;
        }
    }
    __syncthreads();

    // ---- layer 2: out = h @ W_dec ----
    v8f c2[8];
#pragma unroll
    for (int nt = 0; nt < 8; ++nt) c2[nt] = v8f_zero();

#pragma unroll
    for (int kt = 0; kt < 2; ++kt) {
        const __bf16* p = hl + row * HID + kt * 32 + half * 8;
        v8bf lo = *(const v8bf*)(p);
        v8bf hi = *(const v8bf*)(p + 16);
        v16bf a = __builtin_shufflevector(lo, hi,
            0, 1, 2, 3, 4, 5, 6, 7, 8, 9, 10, 11, 12, 13, 14, 15);
#pragma unroll
        for (int nt = 0; nt < 8; ++nt) {
            v16bf bD = *(const v16bf*)(wdec + (size_t)((kt * 8 + nt) * 32 + lane) * 16);
            c2[nt] = wmma_bf16(a, bD, c2[nt]);
        }
    }

    // bias + store (coalesced: 16 consecutive floats per half-wave)
    float* orow = out + (size_t)tile * 16 * IN_CH;
#pragma unroll
    for (int nt = 0; nt < 8; ++nt) {
        int n = nt * 16 + row;
        float bias = bdec[n];
#pragma unroll
        for (int r = 0; r < 8; ++r)
            orow[(size_t)(r + 8 * half) * IN_CH + n] = c2[nt][r] + bias;
    }
}

// ---------------------------------------------------------------------------
// Launch: zero -> weight prep -> edge scatter -> fused WMMA GEMM
// Workspace use: [0, N*128*4) f32 agg ; then 48KB swizzled bf16 weights.
// ---------------------------------------------------------------------------
extern "C" void kernel_launch(void* const* d_in, const int* in_sizes, int n_in,
                              void* d_out, int out_size, void* d_ws, size_t ws_size,
                              hipStream_t stream) {
    const float* x     = (const float*)d_in[0];
    const int*   ei    = (const int*)  d_in[1];
    const float* Wrel  = (const float*)d_in[2];
    const float* brel  = (const float*)d_in[3];
    const float* Wroot = (const float*)d_in[4];
    const float* Wdec  = (const float*)d_in[5];
    const float* bdec  = (const float*)d_in[6];
    float* out = (float*)d_out;

    const int N = in_sizes[0] / IN_CH;   // 50000
    const int E = in_sizes[1] / 2;       // 600000

    float*  agg = (float*)d_ws;
    __bf16* sw  = (__bf16*)((char*)d_ws + (size_t)N * IN_CH * sizeof(float));

    const int n4 = (N * IN_CH) / 4;
    k_zero<<<(n4 + 255) / 256, 256, 0, stream>>>(agg, n4);
    k_prep<<<48, 32, 0, stream>>>(Wrel, Wroot, Wdec, sw);

    const long long st = (long long)E * 32;
    k_scatter<<<(unsigned)((st + 255) / 256), 256, 0, stream>>>(x, ei, agg, E);

    k_gemm<<<N / 16, 32, 0, stream>>>(agg, x, sw, brel, bdec, out);
}
